// transformer_layer_21139829031456
// MI455X (gfx1250) — matrix operand order repaired and checked by reading, not compiled, not run on hardware
//
#include <hip/hip_runtime.h>

typedef __attribute__((ext_vector_type(16))) _Float16 v16h;
typedef __attribute__((ext_vector_type(8)))  _Float16 v8h;
typedef __attribute__((ext_vector_type(8)))  float    v8f;
typedef __attribute__((ext_vector_type(4)))  unsigned int u32x4;
typedef __attribute__((ext_vector_type(4)))  int i32x4;
typedef __attribute__((ext_vector_type(8)))  int i32x8;

#define ACT_NONE 0
#define ACT_RELU 1
#define ACT_SIGMOID 2
#define ACT_TANH 3
#define ACT_GELU 4

#if defined(__gfx1250__) && __has_builtin(__builtin_amdgcn_tensor_load_to_lds) && \
    __has_builtin(__builtin_amdgcn_s_wait_tensorcnt)
#define USE_TDM 1
#else
#define USE_TDM 0
#endif

union HF16 { v16h v; v8h h[2]; };

__device__ __forceinline__ v8f zero8() {
  v8f z;
#pragma unroll
  for (int i = 0; i < 8; ++i) z[i] = 0.f;
  return z;
}

// A-fragment (16x32 f16, row-major source, stride ld halfs):
// lane<16: M=lane, K=0..7 & 16..23 ; lane>=16: M=lane-16, K=8..15 & 24..31
__device__ __forceinline__ v16h ldA(const _Float16* p0, int ld) {
  int lane = threadIdx.x & 31;
  const _Float16* p = p0 + (size_t)(lane & 15) * ld + ((lane >> 4) << 3);
  HF16 f;
  f.h[0] = *(const v8h*)(p);
  f.h[1] = *(const v8h*)(p + 16);
  return f.v;
}

// B-fragment (32x16 f16) from W^T [N][K] (stride ld halfs):
// lane<16: N=lane, K=0..7 & 16..23 ; lane>=16: N=lane-16, K=8..15 & 24..31
__device__ __forceinline__ v16h ldB(const _Float16* p0, int ld) {
  int lane = threadIdx.x & 31;
  const _Float16* p = p0 + (size_t)(lane & 15) * ld + ((lane >> 4) << 3);
  HF16 f;
  f.h[0] = *(const v8h*)(p);
  f.h[1] = *(const v8h*)(p + 16);
  return f.v;
}

__device__ __forceinline__ v8f wmma32(v16h a, v16h b, v8f c) {
  return __builtin_amdgcn_wmma_f32_16x16x32_f16(false, a, false, b, (short)0, c,
                                                false, false);
}

__device__ __forceinline__ float actf(float x, int act) {
  if (act == ACT_RELU) return fmaxf(x, 0.f);
  if (act == ACT_SIGMOID) return 1.f / (1.f + __expf(-x));
  if (act == ACT_TANH) return tanhf(x);
  if (act == ACT_GELU) {
    float x3 = x * x * x;
    return 0.5f * x * (1.f + tanhf(0.7978845608028654f * (x + 0.044715f * x3)));
  }
  return x;
}

// ---------------------------------------------------------------------------
// Generic WMMA GEMM: out = act(A_f16[M,K] @ WT_f16[N,K]^T + bias_scale*bias + addC)
// Block: 256 threads = 8 waves; block tile 128(M) x 64(N); wave tile 16x64.
// Whole 64-row B panel (64 x K halfs, contiguous in W^T) staged into LDS once
// via the Tensor Data Mover (padded to a 520-half row stride to spread rows
// across LDS banks), then the K loop runs barrier-free: ds_load_b128 B frags +
// streamed global A frags + 4 WMMAs per 32-K slice.
// ---------------------------------------------------------------------------
#define LBP_STRIDE 520  // K(512) + 8 halfs pad (16B) per row

__global__ __launch_bounds__(256) void gemm_f16_kernel(
    const _Float16* __restrict__ A, const _Float16* __restrict__ WT,
    const float* __restrict__ bias, float bias_scale,
    const float* __restrict__ addC,
    float* __restrict__ outF, _Float16* __restrict__ outH,
    int N, int K, int act) {
  __shared__ __align__(16) _Float16 lbp[64 * LBP_STRIDE];
  int tiles_n = N >> 6;
  int bm = blockIdx.x / tiles_n;
  int bn = blockIdx.x - bm * tiles_n;
  int wave = threadIdx.x >> 5;
  int lane = threadIdx.x & 31;
  int row0 = bm * 128 + wave * 16;
  int n0 = bn * 64;

#if USE_TDM
  if (threadIdx.x < 32) {  // wave 0 issues the panel DMA (TDM ignores EXEC)
    unsigned long long ga =
        (unsigned long long)(const void*)(WT + (size_t)n0 * K);
    unsigned int lds = (unsigned int)(unsigned long long)(const void*)&lbp[0];
    unsigned int qw = 16u * (unsigned int)K;  // panel size in 8-byte units
    u32x4 g0;
    g0[0] = 1u;                                   // count=1, user descriptor
    g0[1] = lds;                                  // lds_addr
    g0[2] = (unsigned int)(ga & 0xffffffffu);     // global_addr[31:0]
    g0[3] = (unsigned int)((ga >> 32) & 0x1ffffffu) | (2u << 30);  // type=2
    i32x8 g1;
    // data_size=8B, pad_enable, pad_interval=256 DW (1024B), pad_amount=4 DW
    g1[0] = (int)((3u << 16) | (1u << 20) | (7u << 22) | (3u << 25));
    g1[1] = (int)((qw & 0xffffu) << 16);          // tensor_dim0[15:0]
    g1[2] = (int)(((qw >> 16) & 0xffffu) | (1u << 16));  // dim0 hi | dim1=1
    g1[3] = (int)((qw & 0xffffu) << 16);          // tile_dim0
    g1[4] = 1;                                    // tile_dim1 = 1
    g1[5] = (int)qw;                              // tensor_dim0_stride lo
    g1[6] = 0;
    g1[7] = 0;
    i32x4 z4 = {0, 0, 0, 0};
#if __clang_major__ >= 23
    i32x8 z8 = {0, 0, 0, 0, 0, 0, 0, 0};
    __builtin_amdgcn_tensor_load_to_lds(g0, g1, z4, z4, z8, 0);
#else
    __builtin_amdgcn_tensor_load_to_lds(g0, g1, z4, z4, 0);
#endif
    __builtin_amdgcn_s_wait_tensorcnt((short)0);
  }
#else
  {
    const _Float16* src = WT + (size_t)n0 * K;
    int cpr = K >> 3;        // v8h chunks per row
    int chunks = 64 * cpr;
    for (int t = threadIdx.x; t < chunks; t += 256) {
      int r = t / cpr, c = t - r * cpr;
      *(v8h*)&lbp[(size_t)r * LBP_STRIDE + (size_t)c * 8] =
          *(const v8h*)(src + (size_t)t * 8);
    }
  }
#endif
  __syncthreads();

  v8f acc[4];
#pragma unroll
  for (int j = 0; j < 4; ++j) acc[j] = zero8();

  const _Float16* arow = A + (size_t)row0 * K;
  for (int kb = 0; kb < K; kb += 32) {
    if (kb + 32 < K)
      __builtin_prefetch((const void*)(arow + kb + 32), 0, 0);
    v16h af = ldA(arow + kb, K);
#pragma unroll
    for (int j = 0; j < 4; ++j) {
      v16h bf = ldB(&lbp[(size_t)(j * 16) * LBP_STRIDE + kb], LBP_STRIDE);
      acc[j] = wmma32(af, bf, acc[j]);
    }
  }

  int hi = lane >> 4, nn = lane & 15;
#pragma unroll
  for (int j = 0; j < 4; ++j) {
    int col = n0 + j * 16 + nn;
    float bb = bias ? bias_scale * bias[col] : 0.f;
#pragma unroll
    for (int v = 0; v < 8; ++v) {
      size_t idx = (size_t)(row0 + v + hi * 8) * N + col;
      float x = acc[j][v] + bb;
      if (addC) x += addC[idx];
      x = actf(x, act);
      if (outF) outF[idx] = x;
      if (outH) outH[idx] = (_Float16)x;
    }
  }
}

// ---------------------------------------------------------------------------
// LayerNorm over 512-wide rows -> f16
// ---------------------------------------------------------------------------
__global__ __launch_bounds__(256) void ln512_kernel(
    const float* __restrict__ x, const float* __restrict__ sc,
    const float* __restrict__ bs, _Float16* __restrict__ out) {
  __shared__ float rs[256];
  __shared__ float rq[256];
  int row = blockIdx.x;
  int tid = threadIdx.x;
  const float* xr = x + (size_t)row * 512;
  float a = xr[tid], b = xr[tid + 256];
  rs[tid] = a + b;
  rq[tid] = a * a + b * b;
  __syncthreads();
  for (int o = 128; o > 0; o >>= 1) {
    if (tid < o) { rs[tid] += rs[tid + o]; rq[tid] += rq[tid + o]; }
    __syncthreads();
  }
  float mean = rs[0] * (1.f / 512.f);
  float var = rq[0] * (1.f / 512.f) - mean * mean;
  float inv = rsqrtf(var + 1e-6f);
  size_t o = (size_t)row * 512;
  out[o + tid] = (_Float16)((a - mean) * inv * sc[tid] + bs[tid]);
  out[o + tid + 256] = (_Float16)((b - mean) * inv * sc[tid + 256] + bs[tid + 256]);
}

// ---------------------------------------------------------------------------
// Small elementwise / layout kernels
// ---------------------------------------------------------------------------
__global__ void f32_to_f16_kernel(const float* __restrict__ s,
                                  _Float16* __restrict__ d, int n) {
  int i = blockIdx.x * 256 + threadIdx.x;
  if (i < n) d[i] = (_Float16)s[i];
}

__global__ void transpose_f32_to_f16_kernel(const float* __restrict__ s,
                                            _Float16* __restrict__ d, int K, int N) {
  int i = blockIdx.x * 256 + threadIdx.x;  // i = n*K + k
  if (i >= N * K) return;
  int n = i / K, k = i - n * K;
  d[i] = (_Float16)s[(size_t)k * N + n];
}

__global__ void pack_qu_qv_kernel(const float* __restrict__ q,
                                  const float* __restrict__ ub,
                                  const float* __restrict__ vb,
                                  _Float16* __restrict__ qu,
                                  _Float16* __restrict__ qv) {
  int i = blockIdx.x * 256 + threadIdx.x;  // over B*S*512
  int c = i & 511;
  int bs = i >> 9;
  int s = bs & 1023;
  int b = bs >> 10;
  int h = c >> 6;
  int dh = c & 63;
  size_t o = ((size_t)(b * 8 + h) * 1024 + s) * 64 + dh;
  float x = q[i];
  qu[o] = (_Float16)(x + ub[c]);
  qv[o] = (_Float16)(x + vb[c]);
}

__global__ void pack_k_kernel(const float* __restrict__ k, _Float16* __restrict__ kd) {
  int i = blockIdx.x * 256 + threadIdx.x;
  int c = i & 511;
  int bs = i >> 9;
  int s = bs & 1023;
  int b = bs >> 10;
  int h = c >> 6;
  int dh = c & 63;
  size_t o = ((size_t)(b * 8 + h) * 1024 + s) * 64 + dh;
  kd[o] = (_Float16)k[i];
}

__global__ void pack_vT_kernel(const float* __restrict__ v, _Float16* __restrict__ vt) {
  const int S = 1024, SP = 1056;
  int i = blockIdx.x * 256 + threadIdx.x;  // over B*H*64*SP, layout [b,h,dh,s]
  int s = i % SP;
  int r = i / SP;
  int dh = r & 63;
  int bh = r >> 6;
  int h = bh & 7;
  int b = bh >> 3;
  float x = 0.f;
  if (s < S) x = v[((size_t)(b * S + s) * 8 + h) * 64 + dh];
  vt[i] = (_Float16)x;
}

__global__ void pack_r_kernel(const float* __restrict__ rsrc, _Float16* __restrict__ rdst) {
  const int S = 1024, RP = 1040;
  int i = blockIdx.x * 256 + threadIdx.x;  // over 8*RP*64, layout [h,j,dh]
  int dh = i & 63;
  int rest = i >> 6;
  int j = rest % RP;
  int h = rest / RP;
  float x = 0.f;
  if (j < S) x = rsrc[(size_t)j * 512 + h * 64 + dh];
  rdst[i] = (_Float16)x;
}

__global__ void mul_to_f16_kernel(const float* __restrict__ a,
                                  const float* __restrict__ b,
                                  _Float16* __restrict__ o, int n) {
  int i = blockIdx.x * 256 + threadIdx.x;
  if (i < n) o[i] = (_Float16)(a[i] * b[i]);
}

__global__ void gate_combine_kernel(const float* __restrict__ z,
                                    const float* __restrict__ x,
                                    const float* __restrict__ h,
                                    float* __restrict__ outF,
                                    _Float16* __restrict__ reluH, int n) {
  int i = blockIdx.x * 256 + threadIdx.x;
  if (i >= n) return;
  float zz = z[i];
  float o = (1.f - zz) * x[i] + zz * h[i];
  outF[i] = o;
  if (reluH) reluH[i] = (_Float16)fmaxf(o, 0.f);
}

// ---------------------------------------------------------------------------
// Attention with TransformerXL relative shift, flash-style online softmax.
// One wave (32 threads) per 16 query rows of one (b,h).
// bd_shift[q,p] = qv[q] . r[S-1-(q-p)] -> for a 16x16 tile the needed r rows
// form an aligned 32-wide band [jlo, jlo+32); compute two WMMA tiles against
// that band and extract the diagonal with in-half lane permutes.
// ---------------------------------------------------------------------------
__device__ __forceinline__ void score_tile16(
    const _Float16* __restrict__ kB, const _Float16* __restrict__ rB,
    v16h qa0, v16h qa1, v16h va0, v16h va1,
    int p0, int q0, int lane, float* sout) {
  const int S = 1024, DH = 64;
  int hi = lane >> 4, nn = lane & 15;
  v8f ac = zero8();
  ac = wmma32(qa0, ldB(kB + (size_t)p0 * DH, DH), ac);
  ac = wmma32(qa1, ldB(kB + (size_t)p0 * DH + 32, DH), ac);
  int jlo = S - 16 + p0 - q0;  // always >= 0; band zero-padded past S
  v8f g0 = zero8(), g1 = zero8();
  g0 = wmma32(va0, ldB(rB + (size_t)jlo * DH, DH), g0);
  g0 = wmma32(va1, ldB(rB + (size_t)jlo * DH + 32, DH), g0);
  g1 = wmma32(va0, ldB(rB + (size_t)(jlo + 16) * DH, DH), g1);
  g1 = wmma32(va1, ldB(rB + (size_t)(jlo + 16) * DH + 32, DH), g1);
  const float scl = 0.125f;  // 1/sqrt(64)
#pragma unroll
  for (int v = 0; v < 8; ++v) {
    int m = v + hi * 8;           // query row in tile
    int cg = 15 + nn - m;         // band column 0..30
    int srcl = (lane & 16) | (cg & 15);
    float a0 = __shfl(g0[v], srcl, 32);
    float a1 = __shfl(g1[v], srcl, 32);
    float bd = (cg < 16) ? a0 : a1;
    float s = (ac[v] + bd) * scl;
    sout[v] = ((p0 + nn) <= (q0 + m)) ? s : -1e30f;  // causal mask
  }
}

__global__ __launch_bounds__(32) void attn_kernel(
    const _Float16* __restrict__ qu, const _Float16* __restrict__ qv,
    const _Float16* __restrict__ kf, const _Float16* __restrict__ vT,
    const _Float16* __restrict__ rf, _Float16* __restrict__ o16) {
  const int S = 1024, DH = 64, H = 8, SP = S + 32, RP = S + 16;
  int lane = threadIdx.x & 31;
  int qt = blockIdx.x & (S / 16 - 1);
  int bh = blockIdx.x / (S / 16);
  int h = bh & (H - 1);
  int b = bh / H;
  int q0 = qt * 16;

  const _Float16* quB = qu + ((size_t)bh * S + q0) * DH;
  const _Float16* qvB = qv + ((size_t)bh * S + q0) * DH;
  const _Float16* kB = kf + (size_t)bh * S * DH;
  const _Float16* vB = vT + (size_t)bh * DH * SP;
  const _Float16* rB = rf + (size_t)h * RP * DH;

  v16h qa0 = ldA(quB, DH), qa1 = ldA(quB + 32, DH);
  v16h va0 = ldA(qvB, DH), va1 = ldA(qvB + 32, DH);

  v8f oacc[4];
#pragma unroll
  for (int j = 0; j < 4; ++j) oacc[j] = zero8();
  float mrow[8], lrow[8];
#pragma unroll
  for (int v = 0; v < 8; ++v) { mrow[v] = -1e30f; lrow[v] = 0.f; }

  __shared__ __align__(16) _Float16 pl[16 * 32];
  int hi = lane >> 4, nn = lane & 15;

  for (int p0s = 0; p0s <= q0; p0s += 32) {
    float s0[8], s1[8];
    score_tile16(kB, rB, qa0, qa1, va0, va1, p0s, q0, lane, s0);
    if (p0s + 16 <= q0 + 15) {
      score_tile16(kB, rB, qa0, qa1, va0, va1, p0s + 16, q0, lane, s1);
    } else {
#pragma unroll
      for (int v = 0; v < 8; ++v) s1[v] = -1e30f;
    }
    // online softmax over the 32-key block (rows live in 16-lane halves)
#pragma unroll
    for (int v = 0; v < 8; ++v) {
      float rmax = fmaxf(s0[v], s1[v]);
      rmax = fmaxf(rmax, __shfl_xor(rmax, 1, 32));
      rmax = fmaxf(rmax, __shfl_xor(rmax, 2, 32));
      rmax = fmaxf(rmax, __shfl_xor(rmax, 4, 32));
      rmax = fmaxf(rmax, __shfl_xor(rmax, 8, 32));
      float mnew = fmaxf(mrow[v], rmax);
      float corr = __expf(mrow[v] - mnew);
      float e0 = __expf(s0[v] - mnew);
      float e1 = __expf(s1[v] - mnew);
      float es = e0 + e1;
      es += __shfl_xor(es, 1, 32);
      es += __shfl_xor(es, 2, 32);
      es += __shfl_xor(es, 4, 32);
      es += __shfl_xor(es, 8, 32);
      lrow[v] = lrow[v] * corr + es;
      mrow[v] = mnew;
#pragma unroll
      for (int j = 0; j < 4; ++j) oacc[j][v] *= corr;
      int m = v + hi * 8;
      pl[m * 32 + nn] = (_Float16)e0;
      pl[m * 32 + 16 + nn] = (_Float16)e1;
    }
    __syncthreads();
    v16h pa = ldA(pl, 32);  // probs as A-fragment (16x32)
#pragma unroll
    for (int j = 0; j < 4; ++j)
      oacc[j] = wmma32(pa, ldB(vB + (size_t)(j * 16) * SP + p0s, SP), oacc[j]);
    __syncthreads();
  }

#pragma unroll
  for (int v = 0; v < 8; ++v) {
    float inv = 1.f / lrow[v];
    size_t row = (size_t)(b * S + q0 + v + hi * 8);
#pragma unroll
    for (int j = 0; j < 4; ++j)
      o16[row * 512 + h * 64 + j * 16 + nn] = (_Float16)(oacc[j][v] * inv);
  }
}

// ---------------------------------------------------------------------------
// Host orchestration
// ---------------------------------------------------------------------------
extern "C" void kernel_launch(void* const* d_in, const int* in_sizes, int n_in,
                              void* d_out, int out_size, void* d_ws, size_t ws_size,
                              hipStream_t stream) {
  (void)in_sizes; (void)n_in; (void)out_size; (void)ws_size;
  const int B = 8, S = 1024, H = 8, DH = 64;
  const int BS = B * S;                // 8192
  const int NM = BS * 512;             // 4194304
  const int SP = S + 32, RP = S + 16;

  const float* values_keys = (const float*)d_in[0];
  const float* queries     = (const float*)d_in[1];
  const float* pos_embed   = (const float*)d_in[2];
  const float* ln1_s = (const float*)d_in[4];
  const float* ln1_b = (const float*)d_in[5];
  const float* wq = (const float*)d_in[6];  const float* bq = (const float*)d_in[7];
  const float* wk = (const float*)d_in[8];  const float* bk = (const float*)d_in[9];
  const float* wv = (const float*)d_in[10]; const float* bv = (const float*)d_in[11];
  const float* wr = (const float*)d_in[12];
  const float* u_bias = (const float*)d_in[13];
  const float* v_bias = (const float*)d_in[14];
  const float* wo = (const float*)d_in[15]; const float* bo = (const float*)d_in[16];
  const float* ln2_s = (const float*)d_in[17];
  const float* ln2_b = (const float*)d_in[18];
  const float* w1 = (const float*)d_in[19]; const float* b1 = (const float*)d_in[20];
  const float* w2 = (const float*)d_in[21]; const float* b2 = (const float*)d_in[22];
  const float* g1w[6]; for (int i = 0; i < 6; ++i) g1w[i] = (const float*)d_in[23 + i];
  const float* g1_bg = (const float*)d_in[29];
  const float* g2w[6]; for (int i = 0; i < 6; ++i) g2w[i] = (const float*)d_in[30 + i];
  const float* g2_bg = (const float*)d_in[36];

  // workspace carve-out
  char* ws = (char*)d_ws;
  size_t off = 0;
  auto alloc = [&](size_t bytes) -> void* {
    off = (off + 255) & ~(size_t)255;
    void* p = ws + off;
    off += bytes;
    return p;
  };
  float* F0 = (float*)alloc((size_t)NM * 4);
  float* F1 = (float*)alloc((size_t)NM * 4);
  float* F2 = (float*)alloc((size_t)NM * 4);
  float* F3 = (float*)alloc((size_t)NM * 4);
  const size_t HSZ = (size_t)B * H * DH * SP;  // >= NM, covers padded vT
  _Float16* Hb[7];
  for (int i = 0; i < 7; ++i) Hb[i] = (_Float16*)alloc(HSZ * 2);
  _Float16* W16[19];
  for (int i = 0; i < 19; ++i) W16[i] = (_Float16*)alloc((size_t)512 * 512 * 2);
  _Float16* pos16 = (_Float16*)alloc((size_t)S * 512 * 2);
  _Float16* R16 = (_Float16*)alloc((size_t)H * RP * DH * 2);

  // buffer roles (reused across phases)
  _Float16* VKLN = Hb[0];  // -> later O16 -> later X2
  _Float16* QN   = Hb[1];  // -> later Y1
  _Float16* QU   = Hb[2];  // -> later RX
  _Float16* QV   = Hb[3];  // -> later Y2
  _Float16* KF   = Hb[4];  // -> later HLN
  _Float16* VT   = Hb[5];  // -> later FF1
  _Float16* XQ   = Hb[6];  // queries f16
  _Float16* O16 = Hb[0];
  _Float16* Y1  = Hb[1];
  _Float16* RX  = Hb[2];
  _Float16* Y2  = Hb[3];
  _Float16* HLN = Hb[4];
  _Float16* FF1 = Hb[5];
  _Float16* X2  = Hb[0];

  auto gemm = [&](const _Float16* A, const _Float16* WT, const float* bias,
                  float bscale, const float* addC, float* outF, _Float16* outH,
                  int M, int N, int K, int act) {
    int grid = (M / 128) * (N / 64);
    gemm_f16_kernel<<<grid, 256, 0, stream>>>(A, WT, bias, bscale, addC, outF,
                                              outH, N, K, act);
  };

  // 1) weights -> f16 transposed [N][K]
  const float* wsrc[19] = {wq, wk, wv, wr, wo, w1, w2,
                           g1w[0], g1w[1], g1w[2], g1w[3], g1w[4], g1w[5],
                           g2w[0], g2w[1], g2w[2], g2w[3], g2w[4], g2w[5]};
  for (int i = 0; i < 19; ++i)
    transpose_f32_to_f16_kernel<<<(512 * 512) / 256, 256, 0, stream>>>(
        wsrc[i], W16[i], 512, 512);
  f32_to_f16_kernel<<<(S * 512) / 256, 256, 0, stream>>>(pos_embed, pos16, S * 512);
  f32_to_f16_kernel<<<NM / 256, 256, 0, stream>>>(queries, XQ, NM);

  // 2) LN1
  ln512_kernel<<<BS, 256, 0, stream>>>(values_keys, ln1_s, ln1_b, VKLN);
  ln512_kernel<<<BS, 256, 0, stream>>>(queries, ln1_s, ln1_b, QN);

  // 3) Q,K,V,R projections
  gemm(QN, W16[0], bq, 1.f, nullptr, F0, nullptr, BS, 512, 512, ACT_NONE);
  gemm(VKLN, W16[1], bk, 1.f, nullptr, F1, nullptr, BS, 512, 512, ACT_NONE);
  gemm(VKLN, W16[2], bv, 1.f, nullptr, F2, nullptr, BS, 512, 512, ACT_NONE);
  gemm(pos16, W16[3], nullptr, 0.f, nullptr, F3, nullptr, S, 512, 512, ACT_NONE);

  // 4) repack for attention
  pack_qu_qv_kernel<<<NM / 256, 256, 0, stream>>>(F0, u_bias, v_bias, QU, QV);
  pack_k_kernel<<<NM / 256, 256, 0, stream>>>(F1, KF);
  pack_vT_kernel<<<(int)(HSZ / 256), 256, 0, stream>>>(F2, VT);
  pack_r_kernel<<<(H * RP * DH) / 256, 256, 0, stream>>>(F3, R16);

  // 5) attention
  attn_kernel<<<B * H * (S / 16), 32, 0, stream>>>(QU, QV, KF, VT, R16, O16);

  // 6) output projection, y1 = relu(attn_out) in f16
  gemm(O16, W16[4], bo, 1.f, nullptr, nullptr, Y1, BS, 512, 512, ACT_RELU);

  // 7) gate 1: x = queries, y = y1
  gemm(Y1, W16[7], nullptr, 0.f, nullptr, F0, nullptr, BS, 512, 512, ACT_NONE);
  gemm(XQ, W16[8], nullptr, 0.f, F0, F1, nullptr, BS, 512, 512, ACT_SIGMOID);  // r1
  mul_to_f16_kernel<<<NM / 256, 256, 0, stream>>>(F1, queries, RX, NM);        // r1*x
  gemm(Y1, W16[9], nullptr, 0.f, nullptr, F0, nullptr, BS, 512, 512, ACT_NONE);
  gemm(XQ, W16[10], g1_bg, -1.f, F0, F2, nullptr, BS, 512, 512, ACT_SIGMOID);  // z1
  gemm(Y1, W16[11], nullptr, 0.f, nullptr, F0, nullptr, BS, 512, 512, ACT_NONE);
  gemm(RX, W16[12], nullptr, 0.f, F0, F3, nullptr, BS, 512, 512, ACT_TANH);    // h1
  gate_combine_kernel<<<NM / 256, 256, 0, stream>>>(F2, queries, F3, F0, Y2, NM);

  // 8) LN2 + FFN
  ln512_kernel<<<BS, 256, 0, stream>>>(F0, ln2_s, ln2_b, HLN);
  gemm(HLN, W16[5], b1, 1.f, nullptr, nullptr, FF1, BS, 512, 512, ACT_GELU);
  gemm(FF1, W16[6], b2, 1.f, nullptr, F1, X2, BS, 512, 512, ACT_NONE);  // ffn out

  // 9) gate 2: x = ffn out (F1/X2), y = y2 = relu(out_attention)
  gemm(Y2, W16[13], nullptr, 0.f, nullptr, F2, nullptr, BS, 512, 512, ACT_NONE);
  gemm(X2, W16[14], nullptr, 0.f, F2, F3, nullptr, BS, 512, 512, ACT_SIGMOID); // r2
  mul_to_f16_kernel<<<NM / 256, 256, 0, stream>>>(F3, F1, RX, NM);             // r2*x
  gemm(Y2, W16[15], nullptr, 0.f, nullptr, F2, nullptr, BS, 512, 512, ACT_NONE);
  gemm(X2, W16[16], g2_bg, -1.f, F2, F0, nullptr, BS, 512, 512, ACT_SIGMOID);  // z2
  gemm(Y2, W16[17], nullptr, 0.f, nullptr, F2, nullptr, BS, 512, 512, ACT_NONE);
  gemm(RX, W16[18], nullptr, 0.f, F2, F3, nullptr, BS, 512, 512, ACT_TANH);    // h2
  gate_combine_kernel<<<NM / 256, 256, 0, stream>>>(F0, F1, F3, (float*)d_out,
                                                    nullptr, NM);
}
